// VanillaEncoderLayer_51092930953870
// MI455X (gfx1250) — matrix operand order repaired and checked by reading, not compiled, not run on hardware
//
#include <hip/hip_runtime.h>
#include <hip/hip_bf16.h>
#include <math.h>

// ---------------------------------------------------------------------------
// Types / WMMA helpers (CDNA5 gfx1250, wave32)
// ---------------------------------------------------------------------------
typedef __attribute__((ext_vector_type(16))) __bf16        v16bf;
typedef __attribute__((ext_vector_type(8)))  float         v8f;
typedef __attribute__((ext_vector_type(4)))  unsigned int  v4u;
typedef __attribute__((ext_vector_type(8)))  int           v8i;
typedef __attribute__((ext_vector_type(4)))  int           v4i;

union Frag16 { v4u q[2]; v16bf v; __bf16 h[16]; };
union Half8  { v4u q;    __bf16 h[8]; };

#if defined(__has_builtin)
#  if __has_builtin(__builtin_amdgcn_tensor_load_to_lds)
#    define HAVE_TDM 1
#  endif
#endif
#ifndef HAVE_TDM
#  define HAVE_TDM 0
#endif

__device__ __forceinline__ v8f wmma_bf16(v16bf a, v16bf b, v8f c) {
  // D = A(16x32 bf16) * B(32x16 bf16) + C(16x16 f32)
  return __builtin_amdgcn_wmma_f32_16x16x32_bf16(
      /*neg_a=*/false, a, /*neg_b=*/false, b,
      /*c_mod=*/(short)0, c, /*reuse_a=*/false, /*reuse_b=*/false);
}

// A-fragment (16x32 bf16): lane m (= lane&15) holds row m; K contiguous at
// [hi*16, hi*16+16) (matches B-fragment K order)
__device__ __forceinline__ v16bf load_afrag(const __bf16* rowk, int hi) {
  Frag16 f;
  f.q[0] = *(const v4u*)(rowk + hi * 16);
  f.q[1] = *(const v4u*)(rowk + hi * 16 + 8);
  return f.v;
}

// B-fragment (32x16 bf16): lane n (= lane&15) holds column n;
// K = hi*16 .. hi*16+15 are 16 CONTIGUOUS bf16 when data is stored [n][k].
__device__ __forceinline__ v16bf load_bfrag(const __bf16* p) {
  Frag16 f;
  f.q[0] = *(const v4u*)(p);
  f.q[1] = *(const v4u*)(p + 8);
  return f.v;
}

__device__ __forceinline__ v8f zero8() {
  v8f z = {0.f, 0.f, 0.f, 0.f, 0.f, 0.f, 0.f, 0.f};
  return z;
}

// ---------------------------------------------------------------------------
// TDM: async-load a BMxBK bf16 tile (row-major, row stride K elements) from a
// [M,K] tensor into LDS.  Descriptor layout per CDNA5 ISA section 8.3/8.4:
//   group0: [1:0]=count=1, [63:32]=lds_addr, [120:64]=global_addr, [127:126]=2
//   group1: [17:16]=data_size(1=2B), [79:48]=tensor_dim0=K, [111:80]=dim1=M,
//           [127:112]=tile_dim0=BK, [143:128]=tile_dim1=BM,
//           [207:160]=tensor_dim0_stride=K
// Tracked by TENSORcnt; caller must s_wait_tensorcnt then barrier.
// ---------------------------------------------------------------------------
__device__ __forceinline__ void tdm_load_tile_a(const __bf16* gsrc, void* lds_dst,
                                                unsigned K, unsigned M,
                                                unsigned BK, unsigned BM) {
#if HAVE_TDM
  unsigned long long ga = (unsigned long long)(uintptr_t)gsrc;
  unsigned lds_off = (unsigned)(uintptr_t)lds_dst;  // generic ptr low 32b = LDS offset
  v4u g0;
  g0[0] = 1u;                                            // count=1, user descriptor
  g0[1] = lds_off;                                       // LDS byte address
  g0[2] = (unsigned)(ga & 0xFFFFFFFFu);                  // global_addr[31:0]
  g0[3] = (unsigned)((ga >> 32) & 0x01FFFFFFu) | 0x80000000u;  // [120:96] | type=2
  v8i g1;
  g1[0] = (int)(1u << 16);                               // data_size = 2 bytes
  g1[1] = (int)((K & 0xFFFFu) << 16);                    // tensor_dim0 low16
  g1[2] = (int)(((K >> 16) & 0xFFFFu) | ((M & 0xFFFFu) << 16));
  g1[3] = (int)(((M >> 16) & 0xFFFFu) | (BK << 16));     // dim1 hi | tile_dim0
  g1[4] = (int)BM;                                       // tile_dim1 (tile_dim2=0)
  g1[5] = (int)K;                                        // dim0_stride low 32
  g1[6] = 0;
  g1[7] = 0;
  v4i z4 = {0, 0, 0, 0};
#if __clang_major__ >= 23
  v8i z8 = {0, 0, 0, 0, 0, 0, 0, 0};
  __builtin_amdgcn_tensor_load_to_lds(g0, g1, z4, z4, z8, 0);
#else
  __builtin_amdgcn_tensor_load_to_lds(g0, g1, z4, z4, 0);
#endif
#endif
}

// ---------------------------------------------------------------------------
// fp32 -> bf16 conversion (grid-stride)
// ---------------------------------------------------------------------------
__global__ void cvt_f32_bf16_kernel(const float* __restrict__ src,
                                    __bf16* __restrict__ dst, size_t n) {
  size_t i      = (size_t)blockIdx.x * blockDim.x + threadIdx.x;
  size_t stride = (size_t)gridDim.x * blockDim.x;
  for (; i < n; i += stride) dst[i] = (__bf16)src[i];
}

// ---------------------------------------------------------------------------
// Tiled WMMA GEMM:  out[M,N] = A[M,K](bf16) @ B[K,N](bf16) + bias
// Block tile 128x128, BK=32.  256 threads = 8 waves; wave tile 32x64
// (2x4 WMMA subtiles -> 8 v_wmma per K-step per wave).
// A tile staged to LDS by the Tensor Data Mover (fallback: manual b128 copies);
// B tile staged transposed [n][k] so B-fragments are contiguous ds_load_b128.
// EPI: 0 = fp32 store, 1 = bf16 store of (acc+bias)*scale, 2 = bf16 GELU
// ---------------------------------------------------------------------------
template <int EPI>
__global__ __launch_bounds__(256)
void gemm_bf16_kernel(const __bf16* __restrict__ A, const __bf16* __restrict__ Bm,
                      const float* __restrict__ bias, float* __restrict__ outf,
                      __bf16* __restrict__ outb, int M, int N, int K, float scale) {
  __shared__ __bf16 As[128 * 32];  // [m][k], row stride 32
  __shared__ __bf16 Bs[128 * 32];  // transposed: [n][k], row stride 32

  const int tid  = threadIdx.x;
  const int lane = tid & 31;
  const int w    = tid >> 5;       // 0..7
  const int wm   = w >> 1;         // 0..3 -> 32-row strip
  const int wn   = w & 1;          // 0..1 -> 64-col strip
  const int ml   = lane & 15;
  const int hi   = lane >> 4;
  const int bm0  = blockIdx.y * 128;
  const int bn0  = blockIdx.x * 128;

  const int ar = tid >> 1, acol = (tid & 1) * 16;   // manual A loader coords
  const int bk = tid >> 3, bcol = (tid & 7) * 16;   // B loader coords

  v8f acc[2][4];
#pragma unroll
  for (int i = 0; i < 2; ++i)
#pragma unroll
    for (int j = 0; j < 4; ++j) acc[i][j] = zero8();

  for (int k0 = 0; k0 < K; k0 += 32) {
    // ---- stage A tile ----
#if HAVE_TDM
    if (w == 0) {
      tdm_load_tile_a(A + (size_t)bm0 * K + k0, As, (unsigned)K, (unsigned)M,
                      32u, 128u);
    }
#else
    {
      const __bf16* src = A + (size_t)(bm0 + ar) * K + k0 + acol;
      *(v4u*)(As + ar * 32 + acol)     = *(const v4u*)(src);
      *(v4u*)(As + ar * 32 + acol + 8) = *(const v4u*)(src + 8);
      if (k0 + 32 < K) __builtin_prefetch(src + 32, 0, 0);
    }
#endif
    // ---- stage B tile transposed: Bs[n][k] ----
    {
      const __bf16* src = Bm + (size_t)(k0 + bk) * N + bn0 + bcol;
      Half8 u0, u1;
      u0.q = *(const v4u*)(src);
      u1.q = *(const v4u*)(src + 8);
#pragma unroll
      for (int j = 0; j < 8; ++j) {
        Bs[(bcol + j) * 32 + bk]     = u0.h[j];
        Bs[(bcol + 8 + j) * 32 + bk] = u1.h[j];
      }
    }
#if HAVE_TDM
    if (w == 0) __builtin_amdgcn_s_wait_tensorcnt(0);
#endif
    __syncthreads();

    v16bf af[2], bf2[4];
#pragma unroll
    for (int mt = 0; mt < 2; ++mt)
      af[mt] = load_afrag(As + (wm * 32 + mt * 16 + ml) * 32, hi);
#pragma unroll
    for (int nt = 0; nt < 4; ++nt)
      bf2[nt] = load_bfrag(Bs + (wn * 64 + nt * 16 + ml) * 32 + hi * 16);
#pragma unroll
    for (int mt = 0; mt < 2; ++mt)
#pragma unroll
      for (int nt = 0; nt < 4; ++nt)
        acc[mt][nt] = wmma_bf16(af[mt], bf2[nt], acc[mt][nt]);
    __syncthreads();
  }

  // ---- epilogue: C layout is (m = r + 8*hi, n = lane&15) ----
#pragma unroll
  for (int mt = 0; mt < 2; ++mt) {
#pragma unroll
    for (int nt = 0; nt < 4; ++nt) {
      const int col = bn0 + wn * 64 + nt * 16 + ml;
      const float bv = bias[col];
#pragma unroll
      for (int r = 0; r < 8; ++r) {
        const int row = bm0 + wm * 32 + mt * 16 + r + hi * 8;
        float v = acc[mt][nt][r] + bv;
        if (EPI == 0) {
          outf[(size_t)row * N + col] = v;
        } else if (EPI == 1) {
          outb[(size_t)row * N + col] = (__bf16)(v * scale);
        } else {  // exact GELU
          float g = 0.5f * v * (1.0f + erff(v * 0.70710678118654752f));
          outb[(size_t)row * N + col] = (__bf16)g;
        }
      }
    }
  }
}

// ---------------------------------------------------------------------------
// Fused attention: one block per (batch, 16-query-row tile).
// LDS: S[16][2048] f32 | P[16][2048] bf16 | Qs[16][512] bf16 | VT[512][32] bf16
// Q is pre-scaled by 1/sqrt(D) in its GEMM epilogue.
// ---------------------------------------------------------------------------
#define ATTN_N 2048
#define ATTN_D 512
#define ATTN_LDS_BYTES (16*2048*4 + 16*2048*2 + 16*512*2 + 512*32*2 + (256+16+16)*4)

__global__ __launch_bounds__(256)
void attn_kernel(const __bf16* __restrict__ Q, const __bf16* __restrict__ Kmat,
                 const __bf16* __restrict__ Vmat, __bf16* __restrict__ O) {
  extern __shared__ char smem[];
  float*  S    = (float*)smem;                 // 16 x 2048
  __bf16* P    = (__bf16*)(S + 16 * ATTN_N);   // 16 x 2048 (unnormalized exp)
  __bf16* Qs   = P + 16 * ATTN_N;              // 16 x 512
  __bf16* VT   = Qs + 16 * ATTN_D;             // [512][32] transposed V tile
  float*  red  = (float*)(VT + ATTN_D * 32);   // 256
  float*  rmax = red + 256;                    // 16
  float*  rsum = rmax + 16;                    // 16

  const int tid   = threadIdx.x;
  const int lane  = tid & 31;
  const int w     = tid >> 5;   // 0..7
  const int ml    = lane & 15;
  const int hi    = lane >> 4;
  const int batch = blockIdx.y;
  const int q0    = blockIdx.x * 16;

  // ---- load the 16x512 Q tile (rows are contiguous in memory) ----
  {
    const __bf16* qsrc = Q + ((size_t)batch * ATTN_N + q0) * ATTN_D + tid * 32;
#pragma unroll
    for (int j = 0; j < 4; ++j)
      *(v4u*)(Qs + tid * 32 + j * 8) = *(const v4u*)(qsrc + j * 8);
  }
  __syncthreads();

  // ---- Phase 1: S = Q @ K^T (wave w owns key columns [w*256, w*256+256)) ----
  const __bf16* Kbase = Kmat + (size_t)batch * ATTN_N * ATTN_D;
  for (int ct = 0; ct < 16; ++ct) {
    const int col0 = w * 256 + ct * 16;
    v8f acc = zero8();
    for (int k0 = 0; k0 < ATTN_D; k0 += 32) {
      v16bf a = load_afrag(Qs + ml * ATTN_D + k0, hi);
      // K^T fragment: 16 contiguous bf16 of K-row (col0+n) starting at k0+hi*16
      v16bf b = load_bfrag(Kbase + (size_t)(col0 + ml) * ATTN_D + k0 + hi * 16);
      acc = wmma_bf16(a, b, acc);
    }
#pragma unroll
    for (int r = 0; r < 8; ++r)
      S[(r + hi * 8) * ATTN_N + col0 + ml] = acc[r];
  }
  __syncthreads();

  // ---- Phase 2: row softmax (row = tid>>4, each of 16 threads does 128 cols)
  {
    const int row = tid >> 4, seg = tid & 15;
    const float* Srow = S + row * ATTN_N + seg * 128;
    float lmax = -1e30f;
    for (int i = 0; i < 128; ++i) lmax = fmaxf(lmax, Srow[i]);
    red[row * 16 + seg] = lmax;
    __syncthreads();
    if (seg == 0) {
      float m = -1e30f;
      for (int j = 0; j < 16; ++j) m = fmaxf(m, red[row * 16 + j]);
      rmax[row] = m;
    }
    __syncthreads();
    const float mx = rmax[row];
    __bf16* Prow = P + row * ATTN_N + seg * 128;
    float lsum = 0.f;
    for (int i = 0; i < 128; ++i) {
      float e = expf(Srow[i] - mx);
      Prow[i] = (__bf16)e;
      lsum += e;
    }
    red[row * 16 + seg] = lsum;
    __syncthreads();
    if (seg == 0) {
      float s = 0.f;
      for (int j = 0; j < 16; ++j) s += red[row * 16 + j];
      rsum[row] = s;
    }
    __syncthreads();
  }

  // ---- Phase 3: O = P @ V (wave w owns output cols [w*64, w*64+64)) ----
  v8f oacc[4];
#pragma unroll
  for (int ct = 0; ct < 4; ++ct) oacc[ct] = zero8();

  const __bf16* Vbase = Vmat + (size_t)batch * ATTN_N * ATTN_D;
  for (int kv0 = 0; kv0 < ATTN_N; kv0 += 32) {
    // stage V tile transposed: VT[col][k], coalesced 128-bit global reads
    {
      const int kk = tid >> 3, cb = (tid & 7) * 64;
      const __bf16* vsrc = Vbase + (size_t)(kv0 + kk) * ATTN_D + cb;
      Half8 u;
#pragma unroll 2
      for (int c8 = 0; c8 < 8; ++c8) {
        u.q = *(const v4u*)(vsrc + c8 * 8);
#pragma unroll
        for (int j = 0; j < 8; ++j) VT[(cb + c8 * 8 + j) * 32 + kk] = u.h[j];
      }
    }
    __syncthreads();
    v16bf a = load_afrag(P + ml * ATTN_N + kv0, hi);
#pragma unroll
    for (int ct = 0; ct < 4; ++ct) {
      v16bf b = load_bfrag(VT + (w * 64 + ct * 16 + ml) * 32 + hi * 16);
      oacc[ct] = wmma_bf16(a, b, oacc[ct]);
    }
    __syncthreads();
  }

  // ---- epilogue: normalize by 1/rowsum and store bf16 ----
#pragma unroll
  for (int ct = 0; ct < 4; ++ct) {
    const int col = w * 64 + ct * 16 + ml;
#pragma unroll
    for (int r = 0; r < 8; ++r) {
      const int m   = r + hi * 8;
      const float o = oacc[ct][r] / rsum[m];
      O[((size_t)batch * ATTN_N + q0 + m) * ATTN_D + col] = (__bf16)o;
    }
  }
}

// ---------------------------------------------------------------------------
// Residual add + LayerNorm over D=512. One block (256 threads) per row.
// Writes fp32 (always) and bf16 (if outb != nullptr).
// ---------------------------------------------------------------------------
__global__ __launch_bounds__(256)
void add_ln_kernel(const float* __restrict__ a, const float* __restrict__ b,
                   const float* __restrict__ g, const float* __restrict__ be,
                   float* __restrict__ outf, __bf16* __restrict__ outb) {
  __shared__ float red[256];
  __shared__ float sMu, sRstd;
  const int    tid  = threadIdx.x;
  const size_t base = (size_t)blockIdx.x * 512;

  const float v0 = a[base + tid] + b[base + tid];
  const float v1 = a[base + tid + 256] + b[base + tid + 256];

  red[tid] = v0 + v1;
  __syncthreads();
  for (int s = 128; s > 0; s >>= 1) {
    if (tid < s) red[tid] += red[tid + s];
    __syncthreads();
  }
  if (tid == 0) sMu = red[0] * (1.0f / 512.0f);
  __syncthreads();
  const float mu = sMu;
  const float d0 = v0 - mu, d1 = v1 - mu;

  red[tid] = d0 * d0 + d1 * d1;
  __syncthreads();
  for (int s = 128; s > 0; s >>= 1) {
    if (tid < s) red[tid] += red[tid + s];
    __syncthreads();
  }
  if (tid == 0) sRstd = rsqrtf(red[0] * (1.0f / 512.0f) + 1e-5f);
  __syncthreads();
  const float rs = sRstd;

  const float y0 = d0 * rs * g[tid] + be[tid];
  const float y1 = d1 * rs * g[tid + 256] + be[tid + 256];
  outf[base + tid]       = y0;
  outf[base + tid + 256] = y1;
  if (outb) {
    outb[base + tid]       = (__bf16)y0;
    outb[base + tid + 256] = (__bf16)y1;
  }
}

// ---------------------------------------------------------------------------
// Host orchestration
// ---------------------------------------------------------------------------
extern "C" void kernel_launch(void* const* d_in, const int* in_sizes, int n_in,
                              void* d_out, int out_size, void* d_ws, size_t ws_size,
                              hipStream_t stream) {
  (void)in_sizes; (void)n_in; (void)out_size; (void)ws_size;
  const int Bq = 16, Nq = 2048, Dq = 512, Hq = 2048;
  const int M = Bq * Nq;  // 32768

  const float* x  = (const float*)d_in[0];
  const float* wq = (const float*)d_in[1];
  const float* bq = (const float*)d_in[2];
  const float* wk = (const float*)d_in[3];
  const float* bk = (const float*)d_in[4];
  const float* wv = (const float*)d_in[5];
  const float* bv = (const float*)d_in[6];
  const float* wp = (const float*)d_in[7];
  const float* bp = (const float*)d_in[8];
  const float* w1 = (const float*)d_in[9];
  const float* b1 = (const float*)d_in[10];
  const float* w2 = (const float*)d_in[11];
  const float* b2 = (const float*)d_in[12];
  const float* g1 = (const float*)d_in[13];
  const float* be1 = (const float*)d_in[14];
  const float* g2 = (const float*)d_in[15];
  const float* be2 = (const float*)d_in[16];

  // ---- workspace layout (256-byte aligned chunks, with safe aliasing) ----
  char*  wsp = (char*)d_ws;
  size_t off = 0;
  auto alloc = [&](size_t bytes) -> void* {
    void* p = wsp + off;
    off += (bytes + 255) & ~(size_t)255;
    return p;
  };
  __bf16* xb   = (__bf16*)alloc((size_t)M * Dq * 2);   // x bf16; reused as O
  __bf16* wqb  = (__bf16*)alloc((size_t)Dq * Dq * 2);
  __bf16* wkb  = (__bf16*)alloc((size_t)Dq * Dq * 2);
  __bf16* wvb  = (__bf16*)alloc((size_t)Dq * Dq * 2);
  __bf16* wpb  = (__bf16*)alloc((size_t)Dq * Dq * 2);
  __bf16* w1b  = (__bf16*)alloc((size_t)Dq * Hq * 2);
  __bf16* w2b  = (__bf16*)alloc((size_t)Hq * Dq * 2);
  __bf16* Qb   = (__bf16*)alloc((size_t)M * Dq * 2);   // reused as x1 bf16
  __bf16* Kb   = (__bf16*)alloc((size_t)M * Dq * 2);
  __bf16* Vb   = (__bf16*)alloc((size_t)M * Dq * 2);
  float*  attf = (float*)alloc((size_t)M * Dq * 4);    // reused as ff
  float*  x1f  = (float*)alloc((size_t)M * Dq * 4);
  __bf16* hb   = (__bf16*)alloc((size_t)M * Hq * 2);
  __bf16* Ob   = xb;    // alias: x bf16 dead after QKV GEMMs
  __bf16* x1b  = Qb;    // alias: Q dead after attention
  float*  fff  = attf;  // alias: att dead after first LayerNorm

  auto cvt = [&](const float* s, __bf16* d, size_t n) {
    cvt_f32_bf16_kernel<<<4096, 256, 0, stream>>>(s, d, n);
  };

  // 0) bf16 conversions
  cvt(x, xb, (size_t)M * Dq);
  cvt(wq, wqb, (size_t)Dq * Dq);
  cvt(wk, wkb, (size_t)Dq * Dq);
  cvt(wv, wvb, (size_t)Dq * Dq);
  cvt(wp, wpb, (size_t)Dq * Dq);
  cvt(w1, w1b, (size_t)Dq * Hq);
  cvt(w2, w2b, (size_t)Hq * Dq);

  const float qscale = 1.0f / sqrtf((float)Dq);
  dim3 gD(Dq / 128, M / 128);   // N=512 GEMMs
  dim3 gH(Hq / 128, M / 128);   // N=2048 GEMM

  // 1) Q,K,V projections (Q pre-scaled by 1/sqrt(D))
  gemm_bf16_kernel<1><<<gD, 256, 0, stream>>>(xb, wqb, bq, nullptr, Qb, M, Dq, Dq, qscale);
  gemm_bf16_kernel<1><<<gD, 256, 0, stream>>>(xb, wkb, bk, nullptr, Kb, M, Dq, Dq, 1.0f);
  gemm_bf16_kernel<1><<<gD, 256, 0, stream>>>(xb, wvb, bv, nullptr, Vb, M, Dq, Dq, 1.0f);

  // 2) fused attention (softmax(QK^T)V), O in bf16
  attn_kernel<<<dim3(Nq / 16, Bq), 256, ATTN_LDS_BYTES, stream>>>(Qb, Kb, Vb, Ob);

  // 3) output projection (fp32)
  gemm_bf16_kernel<0><<<gD, 256, 0, stream>>>(Ob, wpb, bp, attf, nullptr, M, Dq, Dq, 1.0f);

  // 4) residual + LayerNorm 1 -> x1 (fp32 + bf16)
  add_ln_kernel<<<M, 256, 0, stream>>>(x, attf, g1, be1, x1f, x1b);

  // 5) FFN up-projection + exact GELU -> bf16
  gemm_bf16_kernel<2><<<gH, 256, 0, stream>>>(x1b, w1b, b1, nullptr, hb, M, Hq, Dq, 1.0f);

  // 6) FFN down-projection (fp32)
  gemm_bf16_kernel<0><<<gD, 256, 0, stream>>>(hb, w2b, b2, fff, nullptr, M, Dq, Hq, 1.0f);

  // 7) residual + LayerNorm 2 -> d_out (fp32)
  add_ln_kernel<<<M, 256, 0, stream>>>(x1f, fff, g2, be2, (float*)d_out, nullptr);
}
